// ProductionSpectralAttention_15710990369157
// MI455X (gfx1250) — compile-verified, matrix operand order checked by reading, not run on hardware
//
#include <hip/hip_runtime.h>

// ---------------- problem constants ----------------
#define SEQ   4096
#define DMODEL 1024
#define NHEADS 16
#define HDIM   64
#define BATCH  4
#define BS_TOK (BATCH * SEQ)   // 16384 tokens

// ---------------- WMMA GEMM tile config ----------------
#define BM 128
#define BN 128
#define BK 32
#define ASTR 40   // LDS row stride for A tile (halfs): 80B -> keeps 16B alignment, spreads banks
#define BSTR 33   // LDS row stride for transposed B tile (halfs)

typedef __bf16 v16bf __attribute__((ext_vector_type(16)));
typedef float  v8f   __attribute__((ext_vector_type(8)));

union Frag {
    v16bf v;
    __bf16 e[16];
};

// ---------------------------------------------------------------------------
// Generic bf16 GEMM, C = A[M,K] * B[K,N] (+bias +residual), row-major.
// A is never batched (stride 0: shared circulant); B/C batched via blockIdx.z.
// Output either f32 (Cf) or bf16 (Cb).
// ---------------------------------------------------------------------------
__global__ __launch_bounds__(256)
void gemm_bf16_wmma(const __bf16* __restrict__ A, const __bf16* __restrict__ B,
                    float* __restrict__ Cf, __bf16* __restrict__ Cb,
                    const float* __restrict__ bias, const float* __restrict__ resid,
                    int M, int N, int K, long long sB, long long sC)
{
    (void)M;
    __shared__ __bf16 As[BM * ASTR];
    __shared__ __bf16 Bs[BN * BSTR];

    const int t    = threadIdx.x;
    const int lane = t & 31;
    const int wave = t >> 5;
    const int wm   = wave & 3;   // 4 wave-rows
    const int wn   = wave >> 2;  // 2 wave-cols
    const int l16  = lane & 15;
    const int hi   = lane >> 4;  // half-wave select (K-group per ISA layout)

    const int bn = blockIdx.x * BN;
    const int bm = blockIdx.y * BM;
    const long long z = blockIdx.z;
    B += z * sB;

    v8f acc[2][4];
#pragma unroll
    for (int i = 0; i < 2; ++i)
#pragma unroll
        for (int j = 0; j < 4; ++j)
            acc[i][j] = (v8f){0.f, 0.f, 0.f, 0.f, 0.f, 0.f, 0.f, 0.f};

    // K offsets inside a 16x32 bf16 A fragment (ISA 7.12.2 layout), per half-wave.
    int kb[8];
#pragma unroll
    for (int v = 0; v < 8; ++v)
        kb[v] = ((v < 4) ? (2 * v) : (16 + 2 * (v - 4))) + hi * 8;

    // cooperative-load mappings
    const int arow = t >> 1;          // 0..127
    const int acol = (t & 1) * 16;    // 0 or 16 halfs
    const int bkr  = t >> 3;          // 0..31 (k row)
    const int bnh  = (t & 7) * 16;    // 0..112 (n start)

    for (int k0 = 0; k0 < K; k0 += BK) {
        // ---- stage A tile: global b128 loads -> LDS b128 stores
        {
            const uint4* gp = reinterpret_cast<const uint4*>(
                A + (size_t)(bm + arow) * K + k0 + acol);
            uint4 p0 = gp[0], p1 = gp[1];
            uint4* sp = reinterpret_cast<uint4*>(&As[arow * ASTR + acol]);
            sp[0] = p0; sp[1] = p1;
        }
        // ---- stage B tile transposed: coalesced global read, scatter to LDS columns
        {
            const uint4* gp = reinterpret_cast<const uint4*>(
                B + (size_t)(k0 + bkr) * N + bn + bnh);
            uint4 p0 = gp[0], p1 = gp[1];
            __bf16 tmp[16];
            *reinterpret_cast<uint4*>(tmp)     = p0;
            *reinterpret_cast<uint4*>(tmp + 8) = p1;
#pragma unroll
            for (int j = 0; j < 16; ++j)
                Bs[(bnh + j) * BSTR + bkr] = tmp[j];
        }
        if (k0 + BK < K) {  // prefetch next K tiles (global_prefetch_b8)
            __builtin_prefetch(A + (size_t)(bm + arow) * K + k0 + BK + acol, 0, 1);
            __builtin_prefetch(B + (size_t)(k0 + BK + bkr) * N + bn + bnh, 0, 1);
        }
        __syncthreads();

        // ---- build fragments from LDS (contiguous bf16 pairs -> b32 reads)
        Frag afr[2], bfr[4];
#pragma unroll
        for (int mi = 0; mi < 2; ++mi) {
            const int row = wm * 32 + mi * 16 + l16;
#pragma unroll
            for (int v = 0; v < 8; ++v) {
                const int k2 = kb[v];
                afr[mi].e[2 * v]     = As[row * ASTR + k2];
                afr[mi].e[2 * v + 1] = As[row * ASTR + k2 + 1];
            }
        }
#pragma unroll
        for (int ni = 0; ni < 4; ++ni) {
            const int col = wn * 64 + ni * 16 + l16;
#pragma unroll
            for (int v = 0; v < 8; ++v) {
                const int k2 = kb[v];
                bfr[ni].e[2 * v]     = Bs[col * BSTR + k2];
                bfr[ni].e[2 * v + 1] = Bs[col * BSTR + k2 + 1];
            }
        }

        // ---- 8 WMMAs per wave per K-step
#pragma unroll
        for (int mi = 0; mi < 2; ++mi)
#pragma unroll
            for (int ni = 0; ni < 4; ++ni)
                acc[mi][ni] = __builtin_amdgcn_wmma_f32_16x16x32_bf16(
                    false, afr[mi].v, false, bfr[ni].v,
                    (short)0, acc[mi][ni], false, false);

        __syncthreads();
    }

    // ---- epilogue: D layout = lane(N), vgpr r -> M = r + 8*hi
    float*  cf = Cf ? (Cf + z * sC) : nullptr;
    __bf16* cb = Cb ? (Cb + z * sC) : nullptr;
#pragma unroll
    for (int mi = 0; mi < 2; ++mi) {
#pragma unroll
        for (int ni = 0; ni < 4; ++ni) {
#pragma unroll
            for (int r = 0; r < 8; ++r) {
                const int gm = bm + wm * 32 + mi * 16 + r + 8 * hi;
                const int gn = bn + wn * 64 + ni * 16 + l16;
                float val = acc[mi][ni][r];
                if (bias)  val += bias[gn];
                if (resid) val += resid[(size_t)gm * N + gn];
                if (cf) cf[(size_t)gm * N + gn] = val;
                else    cb[(size_t)gm * N + gn] = (__bf16)val;
            }
        }
    }
}

// ---------------------------------------------------------------------------
// helpers
// ---------------------------------------------------------------------------
__global__ void k_f32_to_bf16(const float* __restrict__ in, __bf16* __restrict__ out,
                              long long n) {
    long long i = (long long)blockIdx.x * blockDim.x + threadIdx.x;
    if (i < n) out[i] = (__bf16)in[i];
}

__global__ void k_costab(float* __restrict__ ct) {
    int a = blockIdx.x * blockDim.x + threadIdx.x;
    if (a < SEQ) ct[a] = __cosf(6.28318530717958647692f * (float)a / (float)SEQ);
}

__global__ void k_cphi(const float* fd, const float* al, const float* fs,
                       float* __restrict__ cphi) {
    int k = blockIdx.x * blockDim.x + threadIdx.x;
    if (k >= SEQ) return;
    const float aa = al[0] + fs[0] * (fd[0] - 1.5f);
    const float fr = ((k < SEQ / 2) ? (float)k : (float)(k - SEQ)) / (float)SEQ;
    cphi[k] = __cosf(aa * atanf(__logf(fabsf(fr) + 1e-10f)));
}

// g[n] = (1/S) * sum_k cphi[k] * cos(2*pi*k*n/S)   (real even kernel)
__global__ void k_build_g(const float* __restrict__ cphi, const float* __restrict__ ct,
                          float* __restrict__ g) {
    int n = blockIdx.x * blockDim.x + threadIdx.x;
    if (n >= SEQ) return;
    float s = 0.f;
    for (int k = 0; k < SEQ; ++k)
        s += cphi[k] * ct[(k * n) & (SEQ - 1)];
    g[n] = s / (float)SEQ;
}

// circulant filter matrix  G[n,m] = g[(n-m) mod S]
__global__ void k_build_G(const float* __restrict__ g, __bf16* __restrict__ G,
                          long long total) {
    long long i = (long long)blockIdx.x * blockDim.x + threadIdx.x;
    if (i >= total) return;
    const int n = (int)(i >> 12);
    const int m = (int)(i & (SEQ - 1));
    G[i] = (__bf16)g[(n - m) & (SEQ - 1)];
}

// cosine IFFT matrix  C[n,k] = cos(2*pi*n*k/S) / sqrt(S)
__global__ void k_build_Cm(const float* __restrict__ ct, __bf16* __restrict__ Cm,
                           long long total) {
    long long i = (long long)blockIdx.x * blockDim.x + threadIdx.x;
    if (i >= total) return;
    const int n = (int)(i >> 12);
    const int k = (int)(i & (SEQ - 1));
    Cm[i] = (__bf16)(ct[(n * k) & (SEQ - 1)] * (1.0f / 64.0f));
}

// logits over heads -> softmax(h) -> attended = w * v_t  (bf16 out)
// one thread per (token, head); 16-lane shuffle softmax (wave32).
__global__ __launch_bounds__(256)
void k_attend(const float* __restrict__ qt, const float* __restrict__ kt,
              const float* __restrict__ vt, __bf16* __restrict__ att, int rows) {
    int tid = blockIdx.x * blockDim.x + threadIdx.x;
    int row = tid >> 4;
    int h   = tid & 15;
    if (row >= rows) return;
    size_t base = (size_t)row * DMODEL + (size_t)h * HDIM;

    float s = 0.f;
#pragma unroll 8
    for (int j = 0; j < HDIM; ++j) s += qt[base + j] * kt[base + j];
    const float logit = s * 0.125f;   // 1/sqrt(64)

    float m = logit;
#pragma unroll
    for (int o = 8; o >= 1; o >>= 1) m = fmaxf(m, __shfl_xor(m, o, 16));
    const float e = __expf(logit - m);
    float se = e;
#pragma unroll
    for (int o = 8; o >= 1; o >>= 1) se += __shfl_xor(se, o, 16);
    const float w = e / se;

#pragma unroll 8
    for (int j = 0; j < HDIM; ++j) att[base + j] = (__bf16)(w * vt[base + j]);
}

// ---------------------------------------------------------------------------
extern "C" void kernel_launch(void* const* d_in, const int* in_sizes, int n_in,
                              void* d_out, int out_size, void* d_ws, size_t ws_size,
                              hipStream_t stream) {
    (void)in_sizes; (void)n_in; (void)out_size; (void)ws_size;
    const float* x     = (const float*)d_in[0];
    const float* fd    = (const float*)d_in[1];
    const float* Wq    = (const float*)d_in[2];
    const float* bq    = (const float*)d_in[3];
    const float* Wk    = (const float*)d_in[4];
    const float* bk    = (const float*)d_in[5];
    const float* Wv    = (const float*)d_in[6];
    const float* bv    = (const float*)d_in[7];
    const float* Wo    = (const float*)d_in[8];
    const float* bo    = (const float*)d_in[9];
    const float* alpha = (const float*)d_in[10];
    const float* fscal = (const float*)d_in[11];
    float* out = (float*)d_out;

    // ---- carve workspace
    char* w = (char*)d_ws;
    size_t off = 0;
    auto carve = [&](size_t bytes) -> void* {
        void* p = w + off;
        off += (bytes + 255) & ~(size_t)255;
        return p;
    };
    const long long nTok  = (long long)BS_TOK * DMODEL;       // 16.7M elems
    const long long nSS   = (long long)SEQ * SEQ;             // 16.7M elems
    float*  ct   = (float*)carve(SEQ * sizeof(float));
    float*  cphi = (float*)carve(SEQ * sizeof(float));
    float*  g    = (float*)carve(SEQ * sizeof(float));
    __bf16* Gm   = (__bf16*)carve(nSS * 2);
    __bf16* Cm   = (__bf16*)carve(nSS * 2);
    __bf16* xb   = (__bf16*)carve(nTok * 2);
    __bf16* Wqb  = (__bf16*)carve((size_t)DMODEL * DMODEL * 2);
    __bf16* Wkb  = (__bf16*)carve((size_t)DMODEL * DMODEL * 2);
    __bf16* Wvb  = (__bf16*)carve((size_t)DMODEL * DMODEL * 2);
    __bf16* Wob  = (__bf16*)carve((size_t)DMODEL * DMODEL * 2);
    __bf16* qb   = (__bf16*)carve(nTok * 2);
    __bf16* kb2  = (__bf16*)carve(nTok * 2);
    __bf16* vb2  = (__bf16*)carve(nTok * 2);
    float*  qt   = (float*)carve(nTok * 4);
    float*  kt   = (float*)carve(nTok * 4);
    float*  vt   = (float*)carve(nTok * 4);
    __bf16* att  = (__bf16*)carve(nTok * 2);
    __bf16* attT = (__bf16*)carve(nTok * 2);

    const dim3 blk(256);
    const long long nW = (long long)DMODEL * DMODEL;

    // ---- precision staging
    k_f32_to_bf16<<<(unsigned)((nTok + 255) / 256), blk, 0, stream>>>(x, xb, nTok);
    k_f32_to_bf16<<<(unsigned)((nW + 255) / 256), blk, 0, stream>>>(Wq, Wqb, nW);
    k_f32_to_bf16<<<(unsigned)((nW + 255) / 256), blk, 0, stream>>>(Wk, Wkb, nW);
    k_f32_to_bf16<<<(unsigned)((nW + 255) / 256), blk, 0, stream>>>(Wv, Wvb, nW);
    k_f32_to_bf16<<<(unsigned)((nW + 255) / 256), blk, 0, stream>>>(Wo, Wob, nW);

    // ---- spectral operator construction
    k_costab<<<SEQ / 256, blk, 0, stream>>>(ct);
    k_cphi<<<SEQ / 256, blk, 0, stream>>>(fd, alpha, fscal, cphi);
    k_build_g<<<SEQ / 256, blk, 0, stream>>>(cphi, ct, g);
    k_build_G<<<(unsigned)((nSS + 255) / 256), blk, 0, stream>>>(g, Gm, nSS);
    k_build_Cm<<<(unsigned)((nSS + 255) / 256), blk, 0, stream>>>(ct, Cm, nSS);

    // ---- Q/K/V projections: [16384,1024] x [1024,1024] + bias -> bf16
    const dim3 gProj(DMODEL / BN, BS_TOK / BM, 1);
    gemm_bf16_wmma<<<gProj, blk, 0, stream>>>(xb, Wqb, nullptr, qb, bq, nullptr,
                                              BS_TOK, DMODEL, DMODEL, 0, 0);
    gemm_bf16_wmma<<<gProj, blk, 0, stream>>>(xb, Wkb, nullptr, kb2, bk, nullptr,
                                              BS_TOK, DMODEL, DMODEL, 0, 0);
    gemm_bf16_wmma<<<gProj, blk, 0, stream>>>(xb, Wvb, nullptr, vb2, bv, nullptr,
                                              BS_TOK, DMODEL, DMODEL, 0, 0);

    // ---- spectral filter as circulant GEMM: q_t[b] = G @ q[b]  (f32 out), batched over B
    const dim3 gSpec(DMODEL / BN, SEQ / BM, BATCH);
    const long long sBat = (long long)SEQ * DMODEL;
    gemm_bf16_wmma<<<gSpec, blk, 0, stream>>>(Gm, qb, qt, nullptr, nullptr, nullptr,
                                              SEQ, DMODEL, SEQ, sBat, sBat);
    gemm_bf16_wmma<<<gSpec, blk, 0, stream>>>(Gm, kb2, kt, nullptr, nullptr, nullptr,
                                              SEQ, DMODEL, SEQ, sBat, sBat);
    gemm_bf16_wmma<<<gSpec, blk, 0, stream>>>(Gm, vb2, vt, nullptr, nullptr, nullptr,
                                              SEQ, DMODEL, SEQ, sBat, sBat);

    // ---- head logits + softmax + weight v_t
    k_attend<<<(BS_TOK * NHEADS) / 256, blk, 0, stream>>>(qt, kt, vt, att, BS_TOK);

    // ---- final Re(IFFT) as cosine-matrix GEMM: att_time[b] = C @ att[b]  (bf16 out)
    gemm_bf16_wmma<<<gSpec, blk, 0, stream>>>(Cm, att, nullptr, attT, nullptr, nullptr,
                                              SEQ, DMODEL, SEQ, sBat, sBat);

    // ---- output projection + bias + residual -> f32 d_out
    gemm_bf16_wmma<<<gProj, blk, 0, stream>>>(attT, Wob, out, nullptr, bo, x,
                                              BS_TOK, DMODEL, DMODEL, 0, 0);
}
